// ELTE_SAA_30657476559104
// MI455X (gfx1250) — compile-verified, matrix operand order checked by reading, not run on hardware
//
#include <hip/hip_runtime.h>
#include <hip/hip_bf16.h>

// ---------------------------------------------------------------------------
// LTE implicit super-resolution, fused for MI455X (gfx1250, wave32, WMMA).
//
// Per wave: M=32 rows = 8 queries x 4 ensemble offsets (two 16-row WMMA
// tiles sharing every B fragment -> 2x operand reuse, 64 FLOP per L2 byte).
//   Stage A: featurize -> LDS x[32][256] f16
//   Stage B: 3-layer MLP via v_wmma_f32_16x16x32_f16 (f16 in / f32 acc),
//            LDS used for the D->A cross-lane transpose between layers.
//   Stage C: in-wave area-weighted reduce over the 4 offsets + bias +
//            bilinear image sample -> out[Q][3] f32.
// d_ws holds only the f16-converted weights (~264 KB).
// ---------------------------------------------------------------------------

typedef _Float16 v16h __attribute__((ext_vector_type(16)));
typedef _Float16 v8h  __attribute__((ext_vector_type(8)));
typedef float    v8f  __attribute__((ext_vector_type(8)));

#define PI_F 3.14159265358979f
#define HW64 64
#define PLANE 4096            // 64*64
#define WAVES_PER_BLOCK 2
#define QS_PER_WAVE 8         // 8 queries x 4 offsets = 32 rows per wave

// Assemble a 16-element WMMA f16 fragment from two contiguous 16-byte chunks
// (16-bit A/B K-packing: elems 0..7 = K in [hi*8, hi*8+7], 8..15 = +16).
__device__ __forceinline__ v16h make_frag(const _Float16* p0) {
    union { v16h v; v8h h[2]; } u;
    u.h[0] = *(const v8h*)(p0);
    u.h[1] = *(const v8h*)(p0 + 16);
    return u.v;
}

// Bilinear grid_sample of the 3x64x64 image, border pad, align_corners=False.
__device__ __forceinline__ float bilin(const float* __restrict__ img,
                                       float cy, float cx, int n) {
    float gy = fminf(fmaxf(((cy + 1.f) * 64.f - 1.f) * 0.5f, 0.f), 63.f);
    float gx = fminf(fmaxf(((cx + 1.f) * 64.f - 1.f) * 0.5f, 0.f), 63.f);
    int y0 = (int)floorf(gy), x0 = (int)floorf(gx);
    int y1 = min(y0 + 1, 63), x1 = min(x0 + 1, 63);
    float wy = gy - (float)y0, wx = gx - (float)x0;
    const float* p = img + n * PLANE;
    return p[y0 * HW64 + x0] * (1.f - wy) * (1.f - wx)
         + p[y0 * HW64 + x1] * (1.f - wy) * wx
         + p[y1 * HW64 + x0] * wy * (1.f - wx)
         + p[y1 * HW64 + x1] * wy * wx;
}

// Convert f32 weights to f16 into the workspace; w3 padded to 16x256.
__global__ __launch_bounds__(256) void lte_convert_weights(
    const float* __restrict__ w1, const float* __restrict__ w2,
    const float* __restrict__ w3, _Float16* __restrict__ wh) {
    int i = blockIdx.x * 256 + threadIdx.x;   // 256 blocks -> i < 65536
    wh[i]         = (_Float16)w1[i];
    wh[65536 + i] = (_Float16)w2[i];
    if (i < 4096) wh[131072 + i] = (i < 768) ? (_Float16)w3[i] : (_Float16)0.f;
}

__global__ __launch_bounds__(WAVES_PER_BLOCK * 32) void lte_fused(
    const float* __restrict__ img,   const float* __restrict__ coef,
    const float* __restrict__ freq,  const float* __restrict__ coord,
    const float* __restrict__ cell,  const float* __restrict__ wcf,
    const float* __restrict__ wph,   const float* __restrict__ b1,
    const float* __restrict__ b2,    const float* __restrict__ b3,
    const _Float16* __restrict__ wh, float* __restrict__ out) {

    __shared__ __align__(16) _Float16 lx[WAVES_PER_BLOCK][32][256];
    __shared__ float lar[WAVES_PER_BLOCK][32];

    const int lane  = threadIdx.x & 31;
    const int wv    = threadIdx.x >> 5;
    const int hi    = lane >> 4;      // half-wave: picks the K sub-range
    const int nl    = lane & 15;      // row (A) / column (B,D) within tile
    const int qbase = (blockIdx.x * WAVES_PER_BLOCK + wv) * QS_PER_WAVE;

    const _Float16* w1h = wh;
    const _Float16* w2h = wh + 65536;
    const _Float16* w3h = wh + 131072;

    // Warm the L2/WGP$ with the (tiny, shared-by-all-waves) f16 weights.
    __builtin_prefetch(w1h + (threadIdx.x << 6), 0, 3);
    __builtin_prefetch(w2h + (threadIdx.x << 6), 0, 3);

    // ---------------- Stage A: featurize 32 rows into LDS ----------------
    const float rxy = 1.0f / 64.0f, eps = 1e-6f;
    for (int m = 0; m < 32; ++m) {
        const int q = qbase + (m >> 2);
        const int o = m & 3;
        const float vx = (o & 2) ? 1.f : -1.f;   // dim-0 (H) offset sign
        const float vy = (o & 1) ? 1.f : -1.f;   // dim-1 (W) offset sign
        const float c0 = coord[q * 2 + 0], c1 = coord[q * 2 + 1];
        const float s0 = fminf(fmaxf(c0 + vx * rxy + eps, -1.f + eps), 1.f - eps);
        const float s1 = fminf(fmaxf(c1 + vy * rxy + eps, -1.f + eps), 1.f - eps);
        const int iy = min(max((int)floorf(((s0 + 1.f) * 64.f - 1.f) * 0.5f + 0.5f), 0), 63);
        const int ix = min(max((int)floorf(((s1 + 1.f) * 64.f - 1.f) * 0.5f + 0.5f), 0), 63);
        const int lin = iy * HW64 + ix;
        const float rel0 = (c0 - (-1.f + (2.f * (float)iy + 1.f) / 64.f)) * 64.f;
        const float rel1 = (c1 - (-1.f + (2.f * (float)ix + 1.f) / 64.f)) * 64.f;
        if (lane == 0) lar[wv][m] = fabsf(rel0 * rel1) + 1e-9f;
        const float rc0 = cell[q * 2 + 0] * 64.f;
        const float rc1 = cell[q * 2 + 1] * 64.f;
        #pragma unroll
        for (int t = 0; t < 4; ++t) {
            const int k = t * 32 + lane;                       // 0..127
            const float s  = rel0 * wcf[k * 2 + 0] + rel1 * wcf[k * 2 + 1];
            const float ph = rc0 * wph[k * 2 + 0] + rc1 * wph[k * 2 + 1];
            const float f  = freq[k * PLANE + lin] * s + ph;
            const float cs = cosf(PI_F * f);
            const float sn = sinf(PI_F * f);
            lx[wv][m][k]       = (_Float16)(coef[k * PLANE + lin] * cs);
            lx[wv][m][k + 128] = (_Float16)(coef[(k + 128) * PLANE + lin] * sn);
        }
    }
    __syncthreads();

    // ---------------- Layer 1: x(32x256) @ w1^T + b1, ReLU ----------------
    v16h xa[8], xb[8];
    #pragma unroll
    for (int kt = 0; kt < 8; ++kt) {
        xa[kt] = make_frag(&lx[wv][nl][kt * 32 + hi * 8]);
        xb[kt] = make_frag(&lx[wv][16 + nl][kt * 32 + hi * 8]);
    }
    __syncthreads();   // all reads of x done; safe to overwrite with h1

    for (int nt = 0; nt < 16; ++nt) {
        v8f acc0 = {}, acc1 = {};
        #pragma unroll
        for (int kt = 0; kt < 8; ++kt) {
            const _Float16* wp = w1h + (nt * 16 + nl) * 256 + kt * 32 + hi * 8;
            v16h bf = make_frag(wp);   // one B fragment feeds two WMMAs
            acc0 = __builtin_amdgcn_wmma_f32_16x16x32_f16(
                false, xa[kt], false, bf, (short)0, acc0, false, false);
            acc1 = __builtin_amdgcn_wmma_f32_16x16x32_f16(
                false, xb[kt], false, bf, (short)0, acc1, false, false);
        }
        const float bias = b1[nt * 16 + nl];
        #pragma unroll
        for (int r = 0; r < 8; ++r) {
            float v0 = acc0[r] + bias;  v0 = v0 > 0.f ? v0 : 0.f;
            float v1 = acc1[r] + bias;  v1 = v1 > 0.f ? v1 : 0.f;
            lx[wv][r + 8 * hi][nt * 16 + nl]      = (_Float16)v0;  // D -> row-major
            lx[wv][16 + r + 8 * hi][nt * 16 + nl] = (_Float16)v1;
        }
    }
    __syncthreads();

    // ---------------- Layer 2: h1 @ w2^T + b2, ReLU ----------------
    v16h ha[8], hb[8];
    #pragma unroll
    for (int kt = 0; kt < 8; ++kt) {
        ha[kt] = make_frag(&lx[wv][nl][kt * 32 + hi * 8]);
        hb[kt] = make_frag(&lx[wv][16 + nl][kt * 32 + hi * 8]);
    }
    __syncthreads();

    for (int nt = 0; nt < 16; ++nt) {
        v8f acc0 = {}, acc1 = {};
        #pragma unroll
        for (int kt = 0; kt < 8; ++kt) {
            const _Float16* wp = w2h + (nt * 16 + nl) * 256 + kt * 32 + hi * 8;
            v16h bf = make_frag(wp);
            acc0 = __builtin_amdgcn_wmma_f32_16x16x32_f16(
                false, ha[kt], false, bf, (short)0, acc0, false, false);
            acc1 = __builtin_amdgcn_wmma_f32_16x16x32_f16(
                false, hb[kt], false, bf, (short)0, acc1, false, false);
        }
        const float bias = b2[nt * 16 + nl];
        #pragma unroll
        for (int r = 0; r < 8; ++r) {
            float v0 = acc0[r] + bias;  v0 = v0 > 0.f ? v0 : 0.f;
            float v1 = acc1[r] + bias;  v1 = v1 > 0.f ? v1 : 0.f;
            lx[wv][r + 8 * hi][nt * 16 + nl]      = (_Float16)v0;
            lx[wv][16 + r + 8 * hi][nt * 16 + nl] = (_Float16)v1;
        }
    }
    __syncthreads();

    // ---------------- Layer 3: h2 @ w3^T (N padded 3->16) ----------------
    v16h g0[8], g1[8];
    #pragma unroll
    for (int kt = 0; kt < 8; ++kt) {
        g0[kt] = make_frag(&lx[wv][nl][kt * 32 + hi * 8]);
        g1[kt] = make_frag(&lx[wv][16 + nl][kt * 32 + hi * 8]);
    }

    v8f p0 = {}, p1 = {};
    #pragma unroll
    for (int kt = 0; kt < 8; ++kt) {
        const _Float16* wp = w3h + nl * 256 + kt * 32 + hi * 8;
        v16h bf = make_frag(wp);
        p0 = __builtin_amdgcn_wmma_f32_16x16x32_f16(
            false, g0[kt], false, bf, (short)0, p0, false, false);
        p1 = __builtin_amdgcn_wmma_f32_16x16x32_f16(
            false, g1[kt], false, bf, (short)0, p1, false, false);
    }

    // ------- Stage C: offset-reduce + bias + bilinear sample + store -------
    // Tile0 D element r -> row m = r + 8*hi (q_local = m>>2, o = m&3);
    // tile1 adds +16 to m.  weight(row m) = area[(m&~3) + (3-o)] / tot.
    const int b0 = 8 * hi;        // tile0 row base
    const int b1r = 16 + 8 * hi;  // tile1 row base
    const float t0 = lar[wv][b0 + 0] + lar[wv][b0 + 1] + lar[wv][b0 + 2] + lar[wv][b0 + 3];
    const float t1 = lar[wv][b0 + 4] + lar[wv][b0 + 5] + lar[wv][b0 + 6] + lar[wv][b0 + 7];
    const float t2 = lar[wv][b1r + 0] + lar[wv][b1r + 1] + lar[wv][b1r + 2] + lar[wv][b1r + 3];
    const float t3 = lar[wv][b1r + 4] + lar[wv][b1r + 5] + lar[wv][b1r + 6] + lar[wv][b1r + 7];
    const float i0 = 1.f / t0, i1 = 1.f / t1, i2 = 1.f / t2, i3 = 1.f / t3;
    float resA = 0.f, resB = 0.f, resC = 0.f, resD = 0.f;
    #pragma unroll
    for (int r = 0; r < 4; ++r) {
        resA += p0[r]     * lar[wv][b0 + (3 - r)]      * i0;
        resB += p0[r + 4] * lar[wv][b0 + 4 + (3 - r)]  * i1;
        resC += p1[r]     * lar[wv][b1r + (3 - r)]     * i2;
        resD += p1[r + 4] * lar[wv][b1r + 4 + (3 - r)] * i3;
    }

    if (nl < 3) {
        const int n = nl;
        const float bb = b3[n];
        const int qA = qbase + 2 * hi;       // tile0, rows 8*hi .. 8*hi+7
        const int qB = qA + 1;
        const int qC = qbase + 4 + 2 * hi;   // tile1
        const int qD = qC + 1;
        out[qA * 3 + n] = resA + bb + bilin(img, coord[qA * 2], coord[qA * 2 + 1], n);
        out[qB * 3 + n] = resB + bb + bilin(img, coord[qB * 2], coord[qB * 2 + 1], n);
        out[qC * 3 + n] = resC + bb + bilin(img, coord[qC * 2], coord[qC * 2 + 1], n);
        out[qD * 3 + n] = resD + bb + bilin(img, coord[qD * 2], coord[qD * 2 + 1], n);
    }
}

extern "C" void kernel_launch(void* const* d_in, const int* in_sizes, int n_in,
                              void* d_out, int out_size, void* d_ws, size_t ws_size,
                              hipStream_t stream) {
    const float* img   = (const float*)d_in[0];   // [1,3,64,64]
    const float* coef  = (const float*)d_in[1];   // [1,256,64,64]
    const float* freq  = (const float*)d_in[2];   // [1,128,64,64]
    const float* coord = (const float*)d_in[3];   // [1,65536,2]
    const float* cell  = (const float*)d_in[4];   // [1,65536,2]
    const float* wcf   = (const float*)d_in[5];   // [128,2]
    const float* wph   = (const float*)d_in[6];   // [128,2]
    const float* w1    = (const float*)d_in[7];   // [256,256]
    const float* b1    = (const float*)d_in[8];
    const float* w2    = (const float*)d_in[9];
    const float* b2    = (const float*)d_in[10];
    const float* w3    = (const float*)d_in[11];  // [3,256]
    const float* b3    = (const float*)d_in[12];
    float* out = (float*)d_out;                   // [1,65536,3] f32

    _Float16* wh = (_Float16*)d_ws;               // 135168 f16 = 264 KB

    lte_convert_weights<<<dim3(256), dim3(256), 0, stream>>>(w1, w2, w3, wh);

    // 65536 queries / (8 per wave * 2 waves per block) = 4096 blocks
    lte_fused<<<dim3(4096), dim3(WAVES_PER_BLOCK * 32), 0, stream>>>(
        img, coef, freq, coord, cell, wcf, wph, b1, b2, b3, wh, out);
}
